// MultiHeadAttention_22445499089016
// MI455X (gfx1250) — matrix operand order repaired and checked by reading, not compiled, not run on hardware
//
#include <hip/hip_runtime.h>
#include <hip/hip_bf16.h>

// GAT multi-head attention for MI455X (gfx1250, wave32, WMMA).
// Pipeline: pack weights (bf16, B-fragment layout) -> QKV proj (WMMA bf16)
//   -> edge scores + segment-max (encoded u32 atomicMax, L2)
//   -> exp + segment-sum (f32 atomicAdd) -> alpha*V aggregation (f32 atomicAdd)
//   -> output projection (WMMA bf16).

#define N_NODES 100000
#define N_EDGES 1600000
#define FDIM    128
#define HEADS   8
#define HDIM    16
#define QSCALE  0.25f   // HDIM^-0.5

typedef __bf16 v16bf __attribute__((ext_vector_type(16)));
typedef float  v8f   __attribute__((ext_vector_type(8)));
typedef unsigned short su8 __attribute__((ext_vector_type(8)));

__device__ __forceinline__ unsigned short f2bf(float x) {
  unsigned b = __float_as_uint(x);
  b += 0x7FFFu + ((b >> 16) & 1u);          // round-to-nearest-even
  return (unsigned short)(b >> 16);
}

// Order-preserving encode of f32 into u32 so atomicMax(u32) == float max.
__device__ __forceinline__ unsigned encf(float x) {
  unsigned b = __float_as_uint(x);
  return (b & 0x80000000u) ? ~b : (b | 0x80000000u);
}
__device__ __forceinline__ float decf(unsigned u) {
  unsigned b = (u & 0x80000000u) ? (u & 0x7FFFFFFFu) : ~u;
  return __uint_as_float(b);
}

__device__ __forceinline__ v16bf load_frag(const unsigned short* p) {
  union { su8 u[2]; v16bf bf; } f;
  f.u[0] = *(const su8*)(p);
  f.u[1] = *(const su8*)(p + 8);
  return f.bf;
}

// ---------------------------------------------------------------------------
// Pack Wq/Wk/Wv/Wo (f32 row-major [128,128], out = x @ W^T) into bf16 in the
// exact WMMA B-fragment lane order: [mat][ntile(8)][kchunk(4)][lane(32)][e(16)].
// B element (lane n, e): N = n&15, K = e<8 ? (n>>4)*8+e : 16+(n>>4)*8+(e-8)  (16-bit B, 32x16).
// Wq (and later bq) pre-scaled by HDIM^-0.5.
// ---------------------------------------------------------------------------
__global__ __launch_bounds__(256)
void gat_pack_weights(const float* __restrict__ Wq, const float* __restrict__ Wk,
                      const float* __restrict__ Wv, const float* __restrict__ Wo,
                      unsigned short* __restrict__ packed) {
  int idx   = blockIdx.x * 256 + threadIdx.x;   // 65536 total
  int mat   = idx >> 14;
  int rem   = idx & 16383;
  int ntile = rem >> 11;
  int rem2  = rem & 2047;
  int chunk = rem2 >> 9;
  int rem3  = rem2 & 511;
  int lane  = rem3 >> 4;
  int e     = rem3 & 15;
  int n  = lane & 15;
  int k  = (e < 8) ? (((lane >> 4) << 3) + e) : (16 + ((lane >> 4) << 3) + (e - 8));
  int nf = ntile * 16 + n;
  int kf = chunk * 32 + k;
  const float* W = (mat == 0) ? Wq : (mat == 1) ? Wk : (mat == 2) ? Wv : Wo;
  float v = W[nf * FDIM + kf];
  if (mat == 0) v *= QSCALE;
  packed[idx] = f2bf(v);
}

__global__ __launch_bounds__(256)
void gat_init_mz(unsigned* __restrict__ m_enc, float* __restrict__ z) {
  int i = blockIdx.x * 256 + threadIdx.x;
  if (i < N_NODES * HEADS) { m_enc[i] = 0x007FFFFFu; /* enc(-inf) */ z[i] = 0.0f; }
}

__global__ __launch_bounds__(256)
void gat_zero_attn(float* __restrict__ attn) {
  attn[blockIdx.x * 256 + threadIdx.x] = 0.0f;
}

// ---------------------------------------------------------------------------
// QKV projection: block = 256 thr = 8 waves handles a 16-node tile.
// Feat tile staged into LDS already in A-fragment order (16-bit A 16x32:
// lane l row = l&15; e<8 -> K=(l>>4)*8+e ; e>=8 -> K=16+(l>>4)*8+(e-8)).
// Wave w owns column tile w; 3 matrices x 4 K-chunks = 12 WMMAs per wave.
// ---------------------------------------------------------------------------
__global__ __launch_bounds__(256)
void gat_qkv_proj(const float* __restrict__ feat,
                  const unsigned short* __restrict__ packedW,
                  const float* __restrict__ bq, const float* __restrict__ bk,
                  const float* __restrict__ bv,
                  float* __restrict__ qb, float* __restrict__ kb,
                  float* __restrict__ vb) {
  __shared__ unsigned short aT[2048];           // 4 chunks * 32 lanes * 16 bf16
  const int tid = threadIdx.x;
  const int rowBase = blockIdx.x * 16;
  for (int i = tid; i < 2048; i += 256) {
    int chunk = i >> 9, rem = i & 511, lane = rem >> 4, e = rem & 15;
    int row = lane & 15, kb8 = (lane >> 4) << 3;
    int kl = (e < 8) ? (kb8 + e) : (16 + kb8 + (e - 8));
    aT[i] = f2bf(feat[(size_t)(rowBase + row) * FDIM + (chunk << 5) + kl]);
  }
  __syncthreads();

  const int lane  = tid & 31;
  const int ntile = tid >> 5;                   // 8 waves -> 8 column tiles
  v16bf a0 = load_frag(&aT[0 * 512 + lane * 16]);
  v16bf a1 = load_frag(&aT[1 * 512 + lane * 16]);
  v16bf a2 = load_frag(&aT[2 * 512 + lane * 16]);
  v16bf a3 = load_frag(&aT[3 * 512 + lane * 16]);
  const int col   = ntile * 16 + (lane & 15);
  const int mbase = (lane >> 4) << 3;

  for (int mat = 0; mat < 3; ++mat) {
    const float* bias = (mat == 0) ? bq : (mat == 1) ? bk : bv;
    float bvv = bias[col];
    if (mat == 0) bvv *= QSCALE;
    v8f acc = {bvv, bvv, bvv, bvv, bvv, bvv, bvv, bvv};
    const unsigned short* wp = packedW + (((mat * 8 + ntile) * 4) << 9) + lane * 16;
    v16bf b0 = load_frag(wp + 0 * 512);
    v16bf b1 = load_frag(wp + 1 * 512);
    v16bf b2 = load_frag(wp + 2 * 512);
    v16bf b3 = load_frag(wp + 3 * 512);
    acc = __builtin_amdgcn_wmma_f32_16x16x32_bf16(false, a0, false, b0, (short)0, acc, false, false);
    acc = __builtin_amdgcn_wmma_f32_16x16x32_bf16(false, a1, false, b1, (short)0, acc, false, false);
    acc = __builtin_amdgcn_wmma_f32_16x16x32_bf16(false, a2, false, b2, (short)0, acc, false, false);
    acc = __builtin_amdgcn_wmma_f32_16x16x32_bf16(false, a3, false, b3, (short)0, acc, false, false);
    float* out = (mat == 0) ? qb : (mat == 1) ? kb : vb;
#pragma unroll
    for (int g = 0; g < 8; ++g)
      out[(size_t)(rowBase + mbase + g) * FDIM + col] = acc[g];
  }
}

// Pass 1: per (edge, head) score = k[src]_h . q[dst]_h ; segment-max via
// order-preserving u32 atomicMax. q/k tables (102 MB) live in the 192 MB L2.
__global__ __launch_bounds__(256)
void gat_edge_scores(const int* __restrict__ esrc, const int* __restrict__ edst,
                     const float* __restrict__ qb, const float* __restrict__ kb,
                     float* __restrict__ sc, unsigned* __restrict__ m_enc) {
  int i = blockIdx.x * 256 + threadIdx.x;       // E*H exact
  int e = i >> 3, h = i & 7;
  __builtin_prefetch(esrc + ((e + 8192) % N_EDGES), 0, 1);
  int src = esrc[e], dst = edst[e];
  const float4* kp = (const float4*)(kb + (size_t)src * FDIM + h * HDIM);
  const float4* qp = (const float4*)(qb + (size_t)dst * FDIM + h * HDIM);
  float s = 0.f;
#pragma unroll
  for (int j = 0; j < 4; ++j) {
    float4 kk = kp[j], qq = qp[j];
    s += kk.x * qq.x + kk.y * qq.y + kk.z * qq.z + kk.w * qq.w;
  }
  sc[i] = s;
  atomicMax(&m_enc[dst * HEADS + h], encf(s));
}

// Pass 2: e = exp(score - m[dst]); z[dst] += e (f32 atomics at L2).
__global__ __launch_bounds__(256)
void gat_edge_expsum(const int* __restrict__ edst, float* __restrict__ sc,
                     const unsigned* __restrict__ m_enc, float* __restrict__ z) {
  int i = blockIdx.x * 256 + threadIdx.x;
  int e = i >> 3, h = i & 7;
  __builtin_prefetch(edst + ((e + 8192) % N_EDGES), 0, 1);
  int dst = edst[e];
  float m = decf(m_enc[dst * HEADS + h]);
  float ex = __expf(sc[i] - m);
  sc[i] = ex;
  atomicAdd(&z[dst * HEADS + h], ex);
}

// Pass 3: attn[dst] += (e/z[dst]) * v[src] ; v + attn (~102 MB) fit in L2.
__global__ __launch_bounds__(256)
void gat_edge_aggregate(const int* __restrict__ esrc, const int* __restrict__ edst,
                        const float* __restrict__ sc, const float* __restrict__ z,
                        const float* __restrict__ vb, float* __restrict__ attn) {
  int i = blockIdx.x * 256 + threadIdx.x;
  int e = i >> 3, h = i & 7;
  __builtin_prefetch(esrc + ((e + 8192) % N_EDGES), 0, 1);
  int src = esrc[e], dst = edst[e];
  float alpha = sc[i] / z[dst * HEADS + h];
  const float4* vp = (const float4*)(vb + (size_t)src * FDIM + h * HDIM);
  float* ap = attn + (size_t)dst * FDIM + h * HDIM;
#pragma unroll
  for (int j = 0; j < 4; ++j) {
    float4 vv = vp[j];
    atomicAdd(ap + 4 * j + 0, alpha * vv.x);
    atomicAdd(ap + 4 * j + 1, alpha * vv.y);
    atomicAdd(ap + 4 * j + 2, alpha * vv.z);
    atomicAdd(ap + 4 * j + 3, alpha * vv.w);
  }
}

// Output projection: identical GEMM structure, matrix slot 3 (Wo), bias bo.
__global__ __launch_bounds__(256)
void gat_out_proj(const float* __restrict__ attn,
                  const unsigned short* __restrict__ packedW,
                  const float* __restrict__ bo, float* __restrict__ out) {
  __shared__ unsigned short aT[2048];
  const int tid = threadIdx.x;
  const int rowBase = blockIdx.x * 16;
  for (int i = tid; i < 2048; i += 256) {
    int chunk = i >> 9, rem = i & 511, lane = rem >> 4, e = rem & 15;
    int row = lane & 15, kb8 = (lane >> 4) << 3;
    int kl = (e < 8) ? (kb8 + e) : (16 + kb8 + (e - 8));
    aT[i] = f2bf(attn[(size_t)(rowBase + row) * FDIM + (chunk << 5) + kl]);
  }
  __syncthreads();

  const int lane  = tid & 31;
  const int ntile = tid >> 5;
  v16bf a0 = load_frag(&aT[0 * 512 + lane * 16]);
  v16bf a1 = load_frag(&aT[1 * 512 + lane * 16]);
  v16bf a2 = load_frag(&aT[2 * 512 + lane * 16]);
  v16bf a3 = load_frag(&aT[3 * 512 + lane * 16]);
  const int col   = ntile * 16 + (lane & 15);
  const int mbase = (lane >> 4) << 3;

  float bvv = bo[col];
  v8f acc = {bvv, bvv, bvv, bvv, bvv, bvv, bvv, bvv};
  const unsigned short* wp = packedW + (((3 * 8 + ntile) * 4) << 9) + lane * 16;
  v16bf b0 = load_frag(wp + 0 * 512);
  v16bf b1 = load_frag(wp + 1 * 512);
  v16bf b2 = load_frag(wp + 2 * 512);
  v16bf b3 = load_frag(wp + 3 * 512);
  acc = __builtin_amdgcn_wmma_f32_16x16x32_bf16(false, a0, false, b0, (short)0, acc, false, false);
  acc = __builtin_amdgcn_wmma_f32_16x16x32_bf16(false, a1, false, b1, (short)0, acc, false, false);
  acc = __builtin_amdgcn_wmma_f32_16x16x32_bf16(false, a2, false, b2, (short)0, acc, false, false);
  acc = __builtin_amdgcn_wmma_f32_16x16x32_bf16(false, a3, false, b3, (short)0, acc, false, false);
#pragma unroll
  for (int g = 0; g < 8; ++g)
    out[(size_t)(rowBase + mbase + g) * FDIM + col] = acc[g];
}

extern "C" void kernel_launch(void* const* d_in, const int* in_sizes, int n_in,
                              void* d_out, int out_size, void* d_ws, size_t ws_size,
                              hipStream_t stream) {
  const float* feat = (const float*)d_in[0];
  const int*   esrc = (const int*)d_in[1];
  const int*   edst = (const int*)d_in[2];
  const float* Wq = (const float*)d_in[3];
  const float* bq = (const float*)d_in[4];
  const float* Wk = (const float*)d_in[5];
  const float* bk = (const float*)d_in[6];
  const float* Wv = (const float*)d_in[7];
  const float* bv = (const float*)d_in[8];
  const float* Wo = (const float*)d_in[9];
  const float* bo = (const float*)d_in[10];
  float* out = (float*)d_out;

  // Workspace layout (all offsets 256B-aligned):
  char* ws = (char*)d_ws;
  unsigned short* packedW = (unsigned short*)ws;               // 131072 B
  float* qb = (float*)(ws + 131072);                           // N*F f32
  float* kb = qb + (size_t)N_NODES * FDIM;
  float* vb = kb + (size_t)N_NODES * FDIM;
  float* sc = vb + (size_t)N_NODES * FDIM;                     // E*H f32
  unsigned* m_enc = (unsigned*)(sc + (size_t)N_EDGES * HEADS); // N*H u32
  float* z = (float*)(m_enc + (size_t)N_NODES * HEADS);        // N*H f32
  float* attn = qb;  // q is dead after edge_scores -> reuse as attn accumulator

  gat_pack_weights<<<256, 256, 0, stream>>>(Wq, Wk, Wv, Wo, packedW);
  gat_init_mz<<<(N_NODES * HEADS + 255) / 256, 256, 0, stream>>>(m_enc, z);
  gat_qkv_proj<<<N_NODES / 16, 256, 0, stream>>>(feat, packedW, bq, bk, bv, qb, kb, vb);
  gat_edge_scores<<<(N_EDGES * HEADS) / 256, 256, 0, stream>>>(esrc, edst, qb, kb, sc, m_enc);
  gat_zero_attn<<<(N_NODES * FDIM) / 256, 256, 0, stream>>>(attn);
  gat_edge_expsum<<<(N_EDGES * HEADS) / 256, 256, 0, stream>>>(edst, sc, m_enc, z);
  gat_edge_aggregate<<<(N_EDGES * HEADS) / 256, 256, 0, stream>>>(esrc, edst, sc, z, vb, attn);
  gat_out_proj<<<N_NODES / 16, 256, 0, stream>>>(attn, packedW, bo, out);
}
